// SelfAttention_56332791054890
// MI455X (gfx1250) — matrix operand order repaired and checked by reading, not compiled, not run on hardware
//
#include <hip/hip_runtime.h>
#include <hip/hip_bf16.h>

typedef _Float16 v16h __attribute__((ext_vector_type(16)));
typedef _Float16 half8 __attribute__((ext_vector_type(8)));
typedef float    v8f  __attribute__((ext_vector_type(8)));

#define C_CH  256
#define C8    32
#define NPIX  4096   // 64*64
#define BATCH 4
#define NCHUNK (NPIX / 32)

// ---------------------------------------------------------------------------
// Kernel 1: q/k projections -> transposed f16: qT[b][n][32], kT[b][m][32]
// ---------------------------------------------------------------------------
__global__ __launch_bounds__(128) void proj_qk_kernel(
    const float* __restrict__ x,
    const float* __restrict__ Wq, const float* __restrict__ bq,
    const float* __restrict__ Wk, const float* __restrict__ bk,
    _Float16* __restrict__ qT, _Float16* __restrict__ kT)
{
    int t = blockIdx.x * 128 + threadIdx.x;       // 0 .. B*N-1
    int b = t / NPIX;
    int n = t % NPIX;
    const float* xp = x + (size_t)b * C_CH * NPIX + n;

    float qa[C8], ka[C8];
#pragma unroll
    for (int o = 0; o < C8; ++o) { qa[o] = bq[o]; ka[o] = bk[o]; }

    for (int c = 0; c < C_CH; ++c) {
        float xv = xp[(size_t)c * NPIX];          // coalesced across lanes (n)
#pragma unroll
        for (int o = 0; o < C8; ++o) {            // Wq/Wk uniform -> s_load
            qa[o] = fmaf(Wq[o * C_CH + c], xv, qa[o]);
            ka[o] = fmaf(Wk[o * C_CH + c], xv, ka[o]);
        }
    }

    _Float16* qrow = qT + (size_t)t * C8;
    _Float16* krow = kT + (size_t)t * C8;
#pragma unroll
    for (int o = 0; o < C8; ++o) {
        qrow[o] = (_Float16)qa[o];
        krow[o] = (_Float16)ka[o];
    }
}

// ---------------------------------------------------------------------------
// Kernel 2: v projection -> vh[b][c][n] (f16, n contiguous)
// ---------------------------------------------------------------------------
__global__ __launch_bounds__(256) void proj_v_kernel(
    const float* __restrict__ x,
    const float* __restrict__ Wv, const float* __restrict__ bv,
    _Float16* __restrict__ vh)
{
    int b  = blockIdx.x >> 4;
    int og = (blockIdx.x & 15) << 4;
    const float* xb = x + (size_t)b * C_CH * NPIX;

    for (int n = threadIdx.x; n < NPIX; n += 256) {
        float acc[16];
#pragma unroll
        for (int o = 0; o < 16; ++o) acc[o] = bv[og + o];
        for (int c = 0; c < C_CH; ++c) {
            float xv = xb[(size_t)c * NPIX + n];
#pragma unroll
            for (int o = 0; o < 16; ++o)
                acc[o] = fmaf(Wv[(og + o) * C_CH + c], xv, acc[o]);
        }
#pragma unroll
        for (int o = 0; o < 16; ++o)
            vh[((size_t)b * C_CH + og + o) * NPIX + n] = (_Float16)acc[o];
    }
}

// ---------------------------------------------------------------------------
// DPP16 butterfly max across each 16-lane half (pure VALU).
// quad_perm(1,0,3,2)=0xB1, quad_perm(2,3,0,1)=0x4E,
// row_half_mirror=0x141, row_mirror=0x140.
// ---------------------------------------------------------------------------
__device__ __forceinline__ float dpp_rowmax16(float x) {
    int t;
    t = __builtin_amdgcn_update_dpp(0, __builtin_bit_cast(int, x), 0xB1, 0xf, 0xf, true);
    x = fmaxf(x, __builtin_bit_cast(float, t));
    t = __builtin_amdgcn_update_dpp(0, __builtin_bit_cast(int, x), 0x4E, 0xf, 0xf, true);
    x = fmaxf(x, __builtin_bit_cast(float, t));
    t = __builtin_amdgcn_update_dpp(0, __builtin_bit_cast(int, x), 0x141, 0xf, 0xf, true);
    x = fmaxf(x, __builtin_bit_cast(float, t));
    t = __builtin_amdgcn_update_dpp(0, __builtin_bit_cast(int, x), 0x140, 0xf, 0xf, true);
    x = fmaxf(x, __builtin_bit_cast(float, t));
    return x;
}

// ---------------------------------------------------------------------------
// Kernel 3: two-pass flash attention. Block = 4 waves sharing (b, channel
// half), each wave owning a 16-row query tile.
//   Pass 1: stream k chunks (async->LDS, double-buffered); 2 WMMA/chunk;
//           per-lane running max only. One DPP16 butterfly at the end gives
//           the exact row max (scores recomputed in pass 2 are bit-identical,
//           so p = exp(s - rmax) <= 1 always: f16-safe, no rescaling).
//   Pass 2: stream k+v chunks; 2 score WMMAs, exp, LDS transpose,
//           1 ones-WMMA row-sum + 8 PV WMMAs per chunk.
// ---------------------------------------------------------------------------
__global__ __launch_bounds__(128) void attn_kernel(
    const _Float16* __restrict__ qT, const _Float16* __restrict__ kT,
    const _Float16* __restrict__ vh,
    const float* __restrict__ x, const float* __restrict__ gamma_p,
    float* __restrict__ out)
{
    __shared__ __align__(128) _Float16 kst[2][32][C8];     // 2 x 2KB
    __shared__ __align__(128) _Float16 vst[2][128][32];    // 2 x 8KB
    __shared__ __align__(64)  _Float16 plds[4][16][32];    // 4KB (per-wave)

    const int w    = threadIdx.x >> 5;
    const int lane = threadIdx.x & 31;
    const int lh   = lane >> 4;
    const int l16  = lane & 15;
    const int koff = 8 * lh;           // unused; A fragments also split K by k0
    const int k0   = 16 * lh;          // B-fragment K split

    const int b     = blockIdx.x >> 7;        // 128 blocks per batch
    const int rem   = blockIdx.x & 127;
    const int ng    = rem >> 1;               // 64-row group
    const int cbase = (rem & 1) * 128;        // channel half (shared by block)
    const int n0    = ng * 64 + w * 16;       // this wave's query rows

    const float gamma = gamma_p[0];

    const _Float16* kbase = kT + (size_t)b * NPIX * C8;
    const _Float16* vbase = vh + (size_t)b * C_CH * NPIX;

    // ---- A fragment: q tile (lane = row n0+l16, halves = K) ----
    union V16 { v16h v; half8 h[2]; };
    V16 aop, pop;
    const _Float16* qrow = qT + (size_t)(b * NPIX + n0 + l16) * C8;
    aop.h[0] = *(const half8*)(qrow + k0);
    aop.h[1] = *(const half8*)(qrow + k0 + 8);

    // cooperative async staging of m-chunk tiles
    auto stage_k = [&](int buf, int m0) {
        const int t = threadIdx.x;                 // 32 rows x 32 halves = 2KB
        const _Float16* g = kbase + (size_t)m0 * C8 + t * 8;
        unsigned l = (unsigned)(uintptr_t)&kst[buf][0][0] + (unsigned)(t * 16);
        asm volatile("global_load_async_to_lds_b128 %0, %1, off"
                     :: "v"(l), "v"(g) : "memory");
    };
    auto stage_v = [&](int buf, int m0) {
#pragma unroll
        for (int j = 0; j < 4; ++j) {              // 128 rows x 32 halves = 8KB
            int qi = threadIdx.x + 128 * j;
            int cl = qi >> 2;
            int mo = (qi & 3) * 8;
            const _Float16* g = vbase + (size_t)(cbase + cl) * NPIX + m0 + mo;
            unsigned l = (unsigned)(uintptr_t)&vst[buf][cl][mo];
            asm volatile("global_load_async_to_lds_b128 %0, %1, off"
                         :: "v"(l), "v"(g) : "memory");
        }
    };

    // ================= pass 1: exact row max =================
    float rmax[8];
#pragma unroll
    for (int r = 0; r < 8; ++r) rmax[r] = -__builtin_inff();

    stage_k(0, 0);
    for (int it = 0; it < NCHUNK; ++it) {
        const int cur = it & 1;
        asm volatile("s_wait_asynccnt 0x0" ::: "memory");
        __syncthreads();
        if (it + 1 < NCHUNK) stage_k(1 - cur, (it + 1) * 32);

        v16h kb0 = *(const v16h*)(&kst[cur][l16][k0]);
        v16h kb1 = *(const v16h*)(&kst[cur][16 + l16][k0]);
        v8f zc = (v8f){};
        v8f sA = __builtin_amdgcn_wmma_f32_16x16x32_f16(
                     false, aop.v, false, kb0, (short)0, zc, false, false);
        v8f sB = __builtin_amdgcn_wmma_f32_16x16x32_f16(
                     false, aop.v, false, kb1, (short)0, zc, false, false);
#pragma unroll
        for (int r = 0; r < 8; ++r)
            rmax[r] = fmaxf(rmax[r], fmaxf(sA[r], sB[r]));
    }
#pragma unroll
    for (int r = 0; r < 8; ++r) rmax[r] = dpp_rowmax16(rmax[r]);

    __syncthreads();   // all waves done reading kst before re-staging buf 0

    // ================= pass 2: probabilities + PV =================
    v16h onesv;
#pragma unroll
    for (int i = 0; i < 16; ++i) onesv[i] = (_Float16)1.0f;

    v8f acc[8], acc_s = (v8f){};
#pragma unroll
    for (int ct = 0; ct < 8; ++ct) acc[ct] = (v8f){};

    stage_k(0, 0);
    stage_v(0, 0);
    for (int it = 0; it < NCHUNK; ++it) {
        const int cur = it & 1;
        asm volatile("s_wait_asynccnt 0x0" ::: "memory");
        __syncthreads();
        if (it + 1 < NCHUNK) {
            stage_k(1 - cur, (it + 1) * 32);
            stage_v(1 - cur, (it + 1) * 32);
        }

        // scores (bit-identical to pass 1)
        v16h kb0 = *(const v16h*)(&kst[cur][l16][k0]);
        v16h kb1 = *(const v16h*)(&kst[cur][16 + l16][k0]);
        v8f zc = (v8f){};
        v8f sA = __builtin_amdgcn_wmma_f32_16x16x32_f16(
                     false, aop.v, false, kb0, (short)0, zc, false, false);
        v8f sB = __builtin_amdgcn_wmma_f32_16x16x32_f16(
                     false, aop.v, false, kb1, (short)0, zc, false, false);

        // probabilities (<=1), transpose C-layout -> A-layout via LDS
#pragma unroll
        for (int r = 0; r < 8; ++r) {
            float pa = __expf(sA[r] - rmax[r]);
            float pb = __expf(sB[r] - rmax[r]);
            plds[w][r + 8 * lh][l16]      = (_Float16)pa;
            plds[w][r + 8 * lh][16 + l16] = (_Float16)pb;
        }
        asm volatile("s_wait_dscnt 0x0" ::: "memory");
        pop.h[0] = *(const half8*)(&plds[w][l16][k0]);
        pop.h[1] = *(const half8*)(&plds[w][l16][k0 + 8]);

        // row-sum tile (B = ones) + PV accumulate from LDS-staged v
        acc_s = __builtin_amdgcn_wmma_f32_16x16x32_f16(
                    false, pop.v, false, onesv, (short)0, acc_s, false, false);
#pragma unroll
        for (int ct = 0; ct < 8; ++ct) {
            v16h vB = *(const v16h*)(&vst[cur][ct * 16 + l16][k0]);
            acc[ct] = __builtin_amdgcn_wmma_f32_16x16x32_f16(
                          false, pop.v, false, vB, (short)0, acc[ct],
                          false, false);
        }
    }

    // ---- epilogue: normalize by WMMA row sums, fuse gamma*out + x ----
#pragma unroll
    for (int r = 0; r < 8; ++r) {
        float inv = 1.0f / acc_s[r];
        int   n   = n0 + r + 8 * lh;
        size_t base = ((size_t)b * NPIX + n) * C_CH;
#pragma unroll
        for (int ct = 0; ct < 8; ++ct) {
            int c = cbase + ct * 16 + l16;
            out[base + c] = gamma * (acc[ct][r] * inv) + x[base + c];
        }
    }
}

// ---------------------------------------------------------------------------
extern "C" void kernel_launch(void* const* d_in, const int* in_sizes, int n_in,
                              void* d_out, int out_size, void* d_ws, size_t ws_size,
                              hipStream_t stream) {
    const float* x     = (const float*)d_in[0];
    const float* Wq    = (const float*)d_in[1];
    const float* bq    = (const float*)d_in[2];
    const float* Wk    = (const float*)d_in[3];
    const float* bk    = (const float*)d_in[4];
    const float* Wv    = (const float*)d_in[5];
    const float* bv    = (const float*)d_in[6];
    const float* gamma = (const float*)d_in[7];
    float* out = (float*)d_out;

    _Float16* qT = (_Float16*)d_ws;                       // 1 MB
    _Float16* kT = qT + (size_t)BATCH * NPIX * C8;        // 1 MB
    _Float16* vh = kT + (size_t)BATCH * NPIX * C8;        // 8 MB

    proj_qk_kernel<<<(BATCH * NPIX) / 128, 128, 0, stream>>>(x, Wq, bq, Wk, bk, qT, kT);
    proj_v_kernel<<<BATCH * (C_CH / 16), 256, 0, stream>>>(x, Wv, bv, vh);
    // blocks: 4 b x 64 row-groups x 2 channel-halves = 512, 4 waves each
    attn_kernel<<<512, 128, 0, stream>>>(qT, kT, vh, x, gamma, out);
}
